// CSOCSSC_v47_42374147342911
// MI455X (gfx1250) — compile-verified, matrix operand order checked by reading, not run on hardware
//
#include <hip/hip_runtime.h>

// ---------------------------------------------------------------------------
// EGNN layer for MI455X (gfx1250): bf16 WMMA GEMMs, 64 edges per workgroup
// (4 M-tiles) so each streamed B-fragment feeds 4 WMMAs (4x less L2 traffic).
// ---------------------------------------------------------------------------

typedef __attribute__((ext_vector_type(16))) __bf16        v16bf;
typedef __attribute__((ext_vector_type(8)))  float         v8f;
typedef __attribute__((ext_vector_type(4)))  unsigned int  v4u;
typedef __attribute__((ext_vector_type(4)))  float         v4f;

#define NODE_DIM 128
#define HID      256
#define EDGE_DIM 32
#define KIN      288          // 2*NODE_DIM + EDGE_DIM = 9 chunks of K=32
#define MSTRIDE  296          // 288 + 8 pad -> conflict-free ds_load_b128
#define HSTRIDE  264          // 256 + 8 pad -> conflict-free ds_load_b128
#define MTILES   4            // M-tiles (of 16 edges) per workgroup
#define EPB      (MTILES*16)  // 64 edges per workgroup

__device__ __forceinline__ unsigned short f2bf(float f) {
  unsigned u = __float_as_uint(f);
  u += 0x7fffu + ((u >> 16) & 1u);        // round-to-nearest-even
  return (unsigned short)(u >> 16);
}
__device__ __forceinline__ float bf2f(unsigned short s) {
  return __uint_as_float(((unsigned)s) << 16);
}
__device__ __forceinline__ float silu(float v) { return v / (1.0f + __expf(-v)); }

// A-fragment (16x32 bf16) per-lane load from LDS row (two b128 reads)
__device__ __forceinline__ v16bf ldsAfrag(const unsigned short* rowPtr, int koff) {
  union { v4u q[2]; v16bf v; } u;
  u.q[0] = *(const v4u*)(rowPtr + koff);
  u.q[1] = *(const v4u*)(rowPtr + koff + 16);
  return u.v;
}
// B-fragment per-lane load from pre-packed global weights (two b128 reads)
__device__ __forceinline__ v16bf gblBfrag(const unsigned short* p) {
  union { v4u q[2]; v16bf v; } u;
  u.q[0] = *(const v4u*)(p);
  u.q[1] = *(const v4u*)(p + 8);
  return u.v;
}

// ---------------------------------------------------------------------------
// out[0:nH) = h ; out[nH:nH+nX) = x   (atomics accumulate on top)
// ---------------------------------------------------------------------------
__global__ void egnn_init_out(const float* __restrict__ h, const float* __restrict__ x,
                              float* __restrict__ out, int nH, int nX) {
  int i = blockIdx.x * blockDim.x + threadIdx.x;
  int stride = gridDim.x * blockDim.x;
  for (int j = i; j < nH; j += stride) out[j] = h[j];
  for (int j = i; j < nX; j += stride) out[nH + j] = x[j];
}

// ---------------------------------------------------------------------------
// Pre-swizzle w_n1 [288,256], w_c1 [288,256], w_n2 [256,128] (row-major
// [K][N] fp32) into bf16 WMMA B-fragments. Fragment f = 512 bf16 (1 KB):
// lane l holds 16 bf16: k = kc*32 + (i>=8?16:0) + (l>>4)*8 + (i&7),
//                        n = nt*16 + (l&15).
// Layout in ws:  [0,144) w_n1 | [144,288) w_c1 | [288,352) w_n2
// ---------------------------------------------------------------------------
__global__ void egnn_pack_weights(const float* __restrict__ w_n1,
                                  const float* __restrict__ w_c1,
                                  const float* __restrict__ w_n2,
                                  unsigned short* __restrict__ pw) {
  int g = blockIdx.x * blockDim.x + threadIdx.x;
  if (g >= 352 * 32) return;
  int f = g >> 5, lane = g & 31;
  const float* W; int N, kc, nt;
  if (f < 144)      { W = w_n1; N = 256; kc = f / 16;         nt = f % 16;         }
  else if (f < 288) { W = w_c1; N = 256; kc = (f - 144) / 16; nt = (f - 144) % 16; }
  else              { W = w_n2; N = 128; kc = (f - 288) / 8;  nt = (f - 288) % 8;  }
  unsigned short* dst = pw + f * 512 + lane * 16;
  int half = lane >> 4, l15 = lane & 15;
#pragma unroll
  for (int i = 0; i < 16; ++i) {
    int k = kc * 32 + ((i >> 3) << 4) + half * 8 + (i & 7);
    int n = nt * 16 + l15;
    dst[i] = f2bf(W[k * N + n]);
  }
}

// ---------------------------------------------------------------------------
// Main fused kernel: one workgroup (8 wave32) per tile of 64 edges.
// ---------------------------------------------------------------------------
__global__ __launch_bounds__(256) void egnn_main(
    const float* __restrict__ h,  const float* __restrict__ x,
    const int*   __restrict__ eSrc, const int* __restrict__ eDst,
    const float* __restrict__ edist,
    const float* __restrict__ w_e1, const float* __restrict__ b_e1,
    const float* __restrict__ w_e2, const float* __restrict__ b_e2,
    const float* __restrict__ b_n1, const float* __restrict__ b_n2,
    const float* __restrict__ b_c1, const float* __restrict__ w_c2,
    const unsigned short* __restrict__ pw,
    float* __restrict__ outH, float* __restrict__ outX, int nEdges)
{
  __shared__ unsigned short sMin[EPB * MSTRIDE];  // m_in tiles, bf16 (37.9 KB)
  __shared__ unsigned short sHN[EPB * HSTRIDE];   // silu(node layer1)  (33.8 KB)
  __shared__ unsigned short sHC[EPB * HSTRIDE];   // silu(coord layer1) (33.8 KB)
  __shared__ int sSrc[EPB], sDst[EPB];

  const int tid  = threadIdx.x;
  const int lane = tid & 31;
  const int wv   = tid >> 5;
  const int half = lane >> 4;
  const int l15  = lane & 15;

  // ---- phase 0a: edge indices for this tile ------------------------------
  if (tid < EPB) {
    int e = blockIdx.x * EPB + tid;
    e = (e < nEdges) ? e : (nEdges - 1);
    sSrc[tid] = eSrc[e];
    sDst[tid] = eDst[e];
  }
  __syncthreads();

  // ---- phase 0b: gather h[src]|h[dst] -> bf16 LDS (L2-resident table) ----
  {
    int cb = (tid & 15) * 16;       // 16-col block 0..240
    int cc = cb & (NODE_DIM - 1);
#pragma unroll
    for (int mt = 0; mt < MTILES; ++mt) {
      int rr = mt * 16 + (tid >> 4);          // edge row 0..63
      int node = (cb < NODE_DIM) ? sSrc[rr] : sDst[rr];
      const v4f* hp = (const v4f*)(h + (size_t)node * NODE_DIM + cc);
      unsigned int pk[8];
#pragma unroll
      for (int q = 0; q < 4; ++q) {
        v4f f = hp[q];
        pk[2 * q + 0] = (unsigned)f2bf(f.x) | ((unsigned)f2bf(f.y) << 16);
        pk[2 * q + 1] = (unsigned)f2bf(f.z) | ((unsigned)f2bf(f.w) << 16);
      }
      v4u* dp = (v4u*)(sMin + rr * MSTRIDE + cb);
      v4u t0; t0.x = pk[0]; t0.y = pk[1]; t0.z = pk[2]; t0.w = pk[3];
      v4u t1; t1.x = pk[4]; t1.y = pk[5]; t1.z = pk[6]; t1.w = pk[7];
      dp[0] = t0; dp[1] = t1;
    }
  }

  // ---- phase 0c: tiny edge MLP -> m_in cols [256..287] -------------------
  if (tid < EPB) {
    int e = blockIdx.x * EPB + tid;
    e = (e < nEdges) ? e : (nEdges - 1);
    float d0 = edist[e];
    float acc[EDGE_DIM];
#pragma unroll
    for (int j = 0; j < EDGE_DIM; ++j) acc[j] = b_e2[j];
    for (int j = 0; j < EDGE_DIM; ++j) {
      float a = silu(fmaf(d0, w_e1[j], b_e1[j]));
      const float* wr = w_e2 + j * EDGE_DIM;
#pragma unroll
      for (int j2 = 0; j2 < EDGE_DIM; ++j2) acc[j2] = fmaf(a, wr[j2], acc[j2]);
    }
    unsigned short* dp = sMin + tid * MSTRIDE + 2 * NODE_DIM;
#pragma unroll
    for (int j2 = 0; j2 < EDGE_DIM; ++j2) dp[j2] = f2bf(acc[j2]);
  }
  __syncthreads();

  const unsigned short* rowA = sMin + l15 * MSTRIDE;   // tile-0 A row
  const unsigned short* pwN1 = pw;
  const unsigned short* pwC1 = pw + 144 * 512;
  const unsigned short* pwN2 = pw + 288 * 512;

  // ---- phase 1: layer-1 GEMMs (node + coord) on WMMA ---------------------
  // hidden[64,256] = silu(m_in[64,288] @ W1[288,256] + b1)
  // wave wv owns N-tiles {2wv, 2wv+1}; each B fragment is reused by 4 M-tiles.
  for (int t = 0; t < 2; ++t) {
    int nt = wv * 2 + t;
    float bn = b_n1[nt * 16 + l15];
    float bc = b_c1[nt * 16 + l15];
    v8f accN[MTILES], accC[MTILES];
#pragma unroll
    for (int mt = 0; mt < MTILES; ++mt) {
      accN[mt] = (v8f){bn, bn, bn, bn, bn, bn, bn, bn};
      accC[mt] = (v8f){bc, bc, bc, bc, bc, bc, bc, bc};
    }
#pragma unroll
    for (int kc = 0; kc < 9; ++kc) {
      if (kc < 8) {
        __builtin_prefetch(pwN1 + ((kc + 1) * 16 + nt) * 512 + lane * 16, 0, 3);
        __builtin_prefetch(pwC1 + ((kc + 1) * 16 + nt) * 512 + lane * 16, 0, 3);
      }
      v16bf bN = gblBfrag(pwN1 + (kc * 16 + nt) * 512 + lane * 16);
      v16bf bC = gblBfrag(pwC1 + (kc * 16 + nt) * 512 + lane * 16);
#pragma unroll
      for (int mt = 0; mt < MTILES; ++mt) {
        v16bf a = ldsAfrag(rowA + mt * 16 * MSTRIDE, kc * 32 + half * 8);
        accN[mt] = __builtin_amdgcn_wmma_f32_16x16x32_bf16(
            false, a, false, bN, (short)0, accN[mt], false, false);
        accC[mt] = __builtin_amdgcn_wmma_f32_16x16x32_bf16(
            false, a, false, bC, (short)0, accC[mt], false, false);
      }
    }
#pragma unroll
    for (int mt = 0; mt < MTILES; ++mt) {
#pragma unroll
      for (int r = 0; r < 8; ++r) {
        int m = mt * 16 + r + half * 8;
        int n = nt * 16 + l15;
        sHN[m * HSTRIDE + n] = f2bf(silu(accN[mt][r]));
        sHC[m * HSTRIDE + n] = f2bf(silu(accC[mt][r]));
      }
    }
  }
  __syncthreads();

  // ---- phase 2: layer-2 node GEMM + scatter-add --------------------------
  // m[64,128] = hiddenN[64,256] @ W2[256,128] + b2; wave wv owns N-tile wv.
  {
    int n0 = wv * 16;
    float b2 = b_n2[n0 + l15];
    v8f acc[MTILES];
#pragma unroll
    for (int mt = 0; mt < MTILES; ++mt)
      acc[mt] = (v8f){b2, b2, b2, b2, b2, b2, b2, b2};
    const unsigned short* rowH = sHN + l15 * HSTRIDE;
#pragma unroll
    for (int kc = 0; kc < 8; ++kc) {
      v16bf b = gblBfrag(pwN2 + (kc * 8 + wv) * 512 + lane * 16);
#pragma unroll
      for (int mt = 0; mt < MTILES; ++mt) {
        v16bf a = ldsAfrag(rowH + mt * 16 * HSTRIDE, kc * 32 + half * 8);
        acc[mt] = __builtin_amdgcn_wmma_f32_16x16x32_bf16(
            false, a, false, b, (short)0, acc[mt], false, false);
      }
    }
#pragma unroll
    for (int mt = 0; mt < MTILES; ++mt) {
#pragma unroll
      for (int r = 0; r < 8; ++r) {
        int m = mt * 16 + r + half * 8;
        atomicAdd(&outH[(size_t)sDst[m] * NODE_DIM + n0 + l15], acc[mt][r]);
      }
    }
  }

  // ---- phase 3: coord head (N=1) + position scatter (waves 0..3) ---------
  if (wv < MTILES) {
    int m = wv * 16 + l15;                      // edge row 0..63
    const unsigned short* hc = sHC + m * HSTRIDE + half * 128;
    const float* wc = w_c2 + half * 128;
    float acc = 0.f;
#pragma unroll 8
    for (int k = 0; k < 128; ++k) acc = fmaf(bf2f(hc[k]), wc[k], acc);
    acc += __shfl_xor(acc, 16, 32);             // combine the two K halves
    if (lane < 16) {
      int s = sSrc[m], d = sDst[m];
      float dx = x[s * 3 + 0] - x[d * 3 + 0];
      float dy = x[s * 3 + 1] - x[d * 3 + 1];
      float dz = x[s * 3 + 2] - x[d * 3 + 2];
      float len = fmaxf(sqrtf(dx * dx + dy * dy + dz * dz), 1e-8f);
      float sc = acc / len;
      atomicAdd(&outX[d * 3 + 0], sc * dx);
      atomicAdd(&outX[d * 3 + 1], sc * dy);
      atomicAdd(&outX[d * 3 + 2], sc * dz);
    }
  }
}

// ---------------------------------------------------------------------------
extern "C" void kernel_launch(void* const* d_in, const int* in_sizes, int n_in,
                              void* d_out, int out_size, void* d_ws, size_t ws_size,
                              hipStream_t stream) {
  (void)n_in; (void)out_size; (void)ws_size;
  const float* h    = (const float*)d_in[0];
  const float* x    = (const float*)d_in[1];
  const int*   ei   = (const int*)d_in[2];     // [2, E]
  const float* ed   = (const float*)d_in[3];
  const float* w_e1 = (const float*)d_in[4];
  const float* b_e1 = (const float*)d_in[5];
  const float* w_e2 = (const float*)d_in[6];
  const float* b_e2 = (const float*)d_in[7];
  const float* w_n1 = (const float*)d_in[8];
  const float* b_n1 = (const float*)d_in[9];
  const float* w_n2 = (const float*)d_in[10];
  const float* b_n2 = (const float*)d_in[11];
  const float* w_c1 = (const float*)d_in[12];
  const float* b_c1 = (const float*)d_in[13];
  const float* w_c2 = (const float*)d_in[14];

  int nH = in_sizes[0];          // N_NODES * 128
  int nX = in_sizes[1];          // N_NODES * 3
  int E  = in_sizes[3];          // N_EDGES

  float* out = (float*)d_out;
  unsigned short* pw = (unsigned short*)d_ws;  // 352 KB of packed bf16 weights

  egnn_init_out<<<1024, 256, 0, stream>>>(h, x, out, nH, nX);
  egnn_pack_weights<<<(352 * 32 + 255) / 256, 256, 0, stream>>>(w_n1, w_c1, w_n2, pw);

  int nTiles = (E + EPB - 1) / EPB;    // 12500 workgroups of 8 wave32
  egnn_main<<<nTiles, 256, 0, stream>>>(h, x, ei, ei + E, ed,
                                        w_e1, b_e1, w_e2, b_e2,
                                        b_n1, b_n2, b_c1, w_c2,
                                        pw, out, out + nH, E);
}
